// InfiniteContextBlock_91319594648399
// MI455X (gfx1250) — compile-verified, matrix operand order checked by reading, not run on hardware
//
#include <hip/hip_runtime.h>
#include <hip/hip_bf16.h>
#include <math.h>

typedef __attribute__((ext_vector_type(16))) _Float16 v16h;
typedef __attribute__((ext_vector_type(8)))  float    v8f;

#define LQ     2048
#define DM     1024
#define NH     16
#define NKVH   4
#define HDIM   64
#define WWIN   256
#define NSTATE 64
#define DTRANK 64
#define SEP    4096
#define CDIM   128
#define DHALF  512
#define XPW    192   // DTRANK + 2*NSTATE

// ---------------------------------------------------------------- LayerNorm
__global__ void ln_kernel(const float* __restrict__ x, const float* __restrict__ g,
                          const float* __restrict__ b, float* __restrict__ xn) {
  __shared__ float red[256];
  int l = blockIdx.x, t = threadIdx.x;
  const float* xr = x + (size_t)l * DM;
  float s = 0.f;
  for (int i = t; i < DM; i += 256) s += xr[i];
  red[t] = s; __syncthreads();
  for (int o = 128; o > 0; o >>= 1) { if (t < o) red[t] += red[t + o]; __syncthreads(); }
  float mu = red[0] / DM;
  __syncthreads();
  float v = 0.f;
  for (int i = t; i < DM; i += 256) { float d = xr[i] - mu; v += d * d; }
  red[t] = v; __syncthreads();
  for (int o = 128; o > 0; o >>= 1) { if (t < o) red[t] += red[t + o]; __syncthreads(); }
  float rstd = rsqrtf(red[0] / DM + 1e-5f);
  for (int i = t; i < DM; i += 256)
    xn[(size_t)l * DM + i] = (xr[i] - mu) * rstd * g[i] + b[i];
}

// ---------------------------------------------------------------- WMMA GEMM
// C[M,N] = epi( A[M,K] * B + bias ), B is K x N (TB=0) or N x K (TB=1).
// EPI: 0=none, 1=+bias, 2=+bias,softplus, 3=+bias,gelu(exact), 4=*scale
// Tiles are staged into LDS with gfx1250 async global->LDS loads (ASYNCcnt).
template <int EPI, bool TB>
__global__ void gemm_wmma(const float* __restrict__ A, const float* __restrict__ B,
                          const float* __restrict__ bias, float* __restrict__ C,
                          int M, int N, int K, int lda, int ldb, int ldc, float scale) {
  __shared__ float Asf[64][32];   // [m][k] f32
  __shared__ float Bsf[64][32];   // [n][k] f32
  int t = threadIdx.x;
  int wave = t >> 5, lane = t & 31;
  int m0 = blockIdx.y * 64, n0 = blockIdx.x * 64;
  int row = lane & 15, kb = (lane >> 4) * 8;

  unsigned long long abase = (unsigned long long)A;
  unsigned long long bbase = (unsigned long long)B;

  v8f acc[4];
  for (int j = 0; j < 4; ++j) for (int i = 0; i < 8; ++i) acc[j][i] = 0.f;

  for (int k0 = 0; k0 < K; k0 += 32) {
    if (k0 + 32 < K)  // prefetch next A tile slice (global_prefetch_b8)
      __builtin_prefetch(&A[(size_t)(m0 + (t >> 1)) * lda + k0 + 32], 0, 1);

    // ---- async stage A tile: 64x32 f32, 16B-vector transfers (4 per thread)
    for (int i = 0; i < 4; ++i) {
      int idx = t * 4 + i;                 // 512 x b128
      int mm = idx >> 3, kk = (idx & 7) * 4;
      unsigned goff = (unsigned)(((size_t)(m0 + mm) * lda + k0 + kk) * sizeof(float));
      unsigned ldst = (unsigned)(unsigned long long)&Asf[mm][kk];
      asm volatile("global_load_async_to_lds_b128 %0, %1, %2"
                   :: "v"(ldst), "v"(goff), "s"(abase) : "memory");
    }
    // ---- async stage B tile into [n][k]
    if (TB) {
      for (int i = 0; i < 4; ++i) {        // N-major, K-contiguous: b128
        int idx = t * 4 + i;
        int nn = idx >> 3, kk = (idx & 7) * 4;
        unsigned goff = (unsigned)(((size_t)(n0 + nn) * ldb + k0 + kk) * sizeof(float));
        unsigned ldst = (unsigned)(unsigned long long)&Bsf[nn][kk];
        asm volatile("global_load_async_to_lds_b128 %0, %1, %2"
                     :: "v"(ldst), "v"(goff), "s"(bbase) : "memory");
      }
    } else {
      for (int i = 0; i < 16; ++i) {       // K x N weights -> transposed LDS: b32
        int idx = t * 16 + i;
        int nn = idx >> 5, kk = idx & 31;
        unsigned goff = (unsigned)(((size_t)(k0 + kk) * ldb + n0 + nn) * sizeof(float));
        unsigned ldst = (unsigned)(unsigned long long)&Bsf[nn][kk];
        asm volatile("global_load_async_to_lds_b32 %0, %1, %2"
                     :: "v"(ldst), "v"(goff), "s"(bbase) : "memory");
      }
    }
    asm volatile("s_wait_asynccnt 0x0" ::: "memory");
    __syncthreads();

    // ---- fragments (f32 LDS -> f16 packs) + WMMA
    v16h a;
    const float* ap = &Asf[wave * 16 + row][0];
    for (int j = 0; j < 8; ++j) a[j]     = (_Float16)ap[kb + j];
    for (int j = 0; j < 8; ++j) a[8 + j] = (_Float16)ap[16 + kb + j];
    for (int j = 0; j < 4; ++j) {
      v16h bf;
      const float* bp = &Bsf[j * 16 + row][0];
      for (int jj = 0; jj < 8; ++jj) bf[jj]     = (_Float16)bp[kb + jj];
      for (int jj = 0; jj < 8; ++jj) bf[8 + jj] = (_Float16)bp[16 + kb + jj];
      acc[j] = __builtin_amdgcn_wmma_f32_16x16x32_f16(false, a, false, bf,
                                                      (short)0, acc[j], false, false);
    }
    __syncthreads();
  }

  int mb = m0 + wave * 16 + (lane >> 4) * 8;
  for (int j = 0; j < 4; ++j) {
    int nn = n0 + j * 16 + (lane & 15);
    float bv = (EPI == 1 || EPI == 2 || EPI == 3) ? bias[nn] : 0.f;
    for (int i = 0; i < 8; ++i) {
      float v = acc[j][i];
      if (EPI == 4) v *= scale;
      v += bv;
      if (EPI == 2) v = (v > 20.f) ? v : log1pf(expf(v));
      if (EPI == 3) v = 0.5f * v * (1.f + erff(v * 0.70710678118f));
      C[(size_t)(mb + i) * ldc + nn] = v;
    }
  }
}

// ---------------------------------------------------------------- SSM scan
__global__ void ssm_scan(const float* __restrict__ delta, const float* __restrict__ xp,
                         const float* __restrict__ xn, const float* __restrict__ A_log,
                         float* __restrict__ ys) {
  __shared__ float Bsh[NSTATE];
  __shared__ float Csh[NSTATE];
  int d = blockIdx.x * 256 + threadIdx.x;
  float h[NSTATE], An[NSTATE];
  for (int n = 0; n < NSTATE; ++n) {
    h[n] = 0.f;
    An[n] = -expf(A_log[(size_t)d * NSTATE + n]);
  }
  for (int tt = 0; tt < LQ; ++tt) {
    int t = threadIdx.x;
    if (t < 64)       Bsh[t]      = xp[(size_t)tt * XPW + DTRANK + t];
    else if (t < 128) Csh[t - 64] = xp[(size_t)tt * XPW + DTRANK + NSTATE + (t - 64)];
    __syncthreads();
    float dt = delta[(size_t)tt * DM + d];
    float dx = dt * xn[(size_t)tt * DM + d];
    float y = 0.f;
    for (int n = 0; n < NSTATE; ++n) {
      h[n] = expf(dt * An[n]) * h[n] + dx * Bsh[n];
      y += h[n] * Csh[n];
    }
    ys[(size_t)tt * DM + d] = y;
    __syncthreads();
  }
}

__global__ void addd_kernel(float* __restrict__ ys, const float* __restrict__ x,
                            const float* __restrict__ Dssm) {
  int i = blockIdx.x * 256 + threadIdx.x;
  ys[i] += x[i] * Dssm[i % DM];
}

// -------------------------------------------------- sliding-window attention
__global__ void attn_kernel(const float* __restrict__ q, const float* __restrict__ k,
                            const float* __restrict__ v, const float* __restrict__ relb,
                            float* __restrict__ ao) {
  __shared__ _Float16 Qs[16][64];    // [q][d]
  __shared__ _Float16 KVs[64][64];   // pass1: [key][d], pass2: [d][key]
  __shared__ float    Sc[16][768];
  int q0 = blockIdx.x * 16;
  int h = blockIdx.y;
  int kvh = h / (NH / NKVH);
  int t = threadIdx.x, wave = t >> 5, lane = t & 31;
  int row = lane & 15, kb = (lane >> 4) * 8;
  int nb = q0 / WWIN;
  int kg0 = (nb - 1) * WWIN;

  for (int i = 0; i < 8; ++i) {
    int idx = t * 8 + i;
    int r = idx >> 6, d = idx & 63;
    Qs[r][d] = (_Float16)q[(size_t)(q0 + r) * DM + h * HDIM + d];
  }
  __syncthreads();

  // scores = Q K^T / 8 + bias, masked
  for (int c = 0; c < 12; ++c) {
    for (int i = 0; i < 32; ++i) {
      int idx = t * 32 + i;
      int j = idx >> 6, d = idx & 63;
      int kg = kg0 + c * 64 + j;
      KVs[j][d] = (kg >= 0 && kg < LQ)
                    ? (_Float16)k[(size_t)kg * (NKVH * HDIM) + kvh * HDIM + d]
                    : (_Float16)0.f;
    }
    __syncthreads();
    v8f acc; for (int i = 0; i < 8; ++i) acc[i] = 0.f;
    for (int s = 0; s < 2; ++s) {
      v16h a, bf;
      const _Float16* ap = &Qs[row][s * 32];
      ((float4*)&a)[0] = *(const float4*)(ap + kb);
      ((float4*)&a)[1] = *(const float4*)(ap + 16 + kb);
      const _Float16* bp = &KVs[wave * 16 + row][s * 32];
      ((float4*)&bf)[0] = *(const float4*)(bp + kb);
      ((float4*)&bf)[1] = *(const float4*)(bp + 16 + kb);
      acc = __builtin_amdgcn_wmma_f32_16x16x32_f16(false, a, false, bf,
                                                   (short)0, acc, false, false);
    }
    int keyl = c * 64 + wave * 16 + (lane & 15);
    int kg = kg0 + keyl;
    for (int i = 0; i < 8; ++i) {
      int ql = (lane >> 4) * 8 + i;
      int l = q0 + ql;
      int rel = kg - l;
      bool valid = (kg >= 0) && (kg < LQ) && (rel >= -WWIN) && (rel <= WWIN);
      float val;
      if (valid) {
        int ri = rel + WWIN;  // [0, 2W]
        val = acc[i] * 0.125f + relb[(size_t)ri * NH + h];
      } else val = -1.0e9f;
      Sc[ql][keyl] = val;
    }
    __syncthreads();
  }

  if (t < 16) {
    float m = -1.0e30f;
    for (int j = 0; j < 768; ++j) m = fmaxf(m, Sc[t][j]);
    float s = 0.f;
    for (int j = 0; j < 768; ++j) { float e = expf(Sc[t][j] - m); Sc[t][j] = e; s += e; }
    float inv = 1.f / s;
    for (int j = 0; j < 768; ++j) Sc[t][j] *= inv;
  }
  __syncthreads();

  // out = P V
  v8f acc; for (int i = 0; i < 8; ++i) acc[i] = 0.f;
  for (int c = 0; c < 12; ++c) {
    for (int i = 0; i < 32; ++i) {
      int idx = t * 32 + i;
      int j = idx >> 6, d = idx & 63;
      int kg = kg0 + c * 64 + j;
      KVs[d][j] = (kg >= 0 && kg < LQ)
                    ? (_Float16)v[(size_t)kg * (NKVH * HDIM) + kvh * HDIM + d]
                    : (_Float16)0.f;
    }
    __syncthreads();
    for (int s = 0; s < 2; ++s) {
      v16h a, bf;
      int kk = c * 64 + s * 32;
      for (int j = 0; j < 8; ++j) a[j]     = (_Float16)Sc[row][kk + kb + j];
      for (int j = 0; j < 8; ++j) a[8 + j] = (_Float16)Sc[row][kk + 16 + kb + j];
      const _Float16* bp = &KVs[wave * 16 + row][s * 32];
      ((float4*)&bf)[0] = *(const float4*)(bp + kb);
      ((float4*)&bf)[1] = *(const float4*)(bp + 16 + kb);
      acc = __builtin_amdgcn_wmma_f32_16x16x32_f16(false, a, false, bf,
                                                   (short)0, acc, false, false);
    }
    __syncthreads();
  }
  for (int i = 0; i < 8; ++i) {
    int ql = (lane >> 4) * 8 + i;
    int d = wave * 16 + (lane & 15);
    ao[(size_t)(q0 + ql) * DM + h * HDIM + d] = acc[i];
  }
}

// ---------------------------------------------------------------- conv+SiLU
__global__ void conv_kernel(const float* __restrict__ x, const float* __restrict__ wconv,
                            const float* __restrict__ bconv, float* __restrict__ out) {
  int idx = blockIdx.x * 256 + threadIdx.x;
  int l = idx / DM, d = idx % DM;
  float acc = bconv[d];
  for (int j = 0; j < 4; ++j) {
    int lt = l - 3 + j;
    if (lt >= 0) acc += x[(size_t)lt * DM + d] * wconv[d * 4 + j];
  }
  out[idx] = acc / (1.f + expf(-acc));
}

// ------------------------------------------------------------ es softmax row
__global__ void softmax_rows(float* __restrict__ es) {
  __shared__ float red[256];
  int l = blockIdx.x, t = threadIdx.x;
  float* r = es + (size_t)l * SEP;
  float m = -1.0e30f;
  for (int i = t; i < SEP; i += 256) m = fmaxf(m, r[i]);
  red[t] = m; __syncthreads();
  for (int o = 128; o > 0; o >>= 1) { if (t < o) red[t] = fmaxf(red[t], red[t + o]); __syncthreads(); }
  m = red[0]; __syncthreads();
  float s = 0.f;
  for (int i = t; i < SEP; i += 256) { float e = expf(r[i] - m); r[i] = e; s += e; }
  red[t] = s; __syncthreads();
  for (int o = 128; o > 0; o >>= 1) { if (t < o) red[t] += red[t + o]; __syncthreads(); }
  float inv = 1.f / red[0];
  for (int i = t; i < SEP; i += 256) r[i] *= inv;
}

// ---------------------------------------------------------------- combine
__global__ void combine_kernel(const float* __restrict__ x, const float* __restrict__ attn,
                               const float* __restrict__ ssm, const float* __restrict__ conv,
                               const float* __restrict__ wm, const float* __restrict__ eo,
                               float* __restrict__ out) {
  int i = blockIdx.x * 256 + threadIdx.x;
  out[i] = x[i] + 0.2f * attn[i] + 0.45f * ssm[i] + 0.15f * conv[i]
               + 0.2f * (wm[i] + 0.3f * eo[i]);
}

// ---------------------------------------------------------------- launcher
#define GEMM(EPI, TB, Ap, Bp, biasp, Cp, M, N, K, lda, ldb, ldc, sc) \
  gemm_wmma<EPI, TB><<<dim3((N) / 64, (M) / 64), 128, 0, stream>>>(  \
      Ap, Bp, biasp, Cp, M, N, K, lda, ldb, ldc, sc)

extern "C" void kernel_launch(void* const* d_in, const int* in_sizes, int n_in,
                              void* d_out, int out_size, void* d_ws, size_t ws_size,
                              hipStream_t stream) {
  (void)in_sizes; (void)n_in; (void)out_size; (void)ws_size;
  const float* x        = (const float*)d_in[0];
  const float* ln_g     = (const float*)d_in[1];
  const float* ln_b     = (const float*)d_in[2];
  const float* w_xproj  = (const float*)d_in[3];
  const float* w_dt     = (const float*)d_in[4];
  const float* b_dt     = (const float*)d_in[5];
  const float* A_log    = (const float*)d_in[6];
  const float* D_ssm    = (const float*)d_in[7];
  const float* w_ssmout = (const float*)d_in[8];
  const float* wq       = (const float*)d_in[9];
  const float* wk       = (const float*)d_in[10];
  const float* wv       = (const float*)d_in[11];
  const float* wo       = (const float*)d_in[12];
  const float* rel_bias = (const float*)d_in[13];
  const float* w_wm     = (const float*)d_in[14];
  const float* b_wm     = (const float*)d_in[15];
  const float* episodic = (const float*)d_in[16];
  const float* w_dc1    = (const float*)d_in[17];
  const float* b_dc1    = (const float*)d_in[18];
  const float* w_dc2    = (const float*)d_in[19];
  const float* b_dc2    = (const float*)d_in[20];
  const float* w_conv   = (const float*)d_in[23];
  const float* b_conv   = (const float*)d_in[24];
  float* out = (float*)d_out;

  float* ws = (float*)d_ws;
  size_t o = 0;
  float* xn      = ws + o; o += (size_t)LQ * DM;
  float* xp      = ws + o; o += (size_t)LQ * XPW;
  float* delta   = ws + o; o += (size_t)LQ * DM;
  float* ys      = ws + o; o += (size_t)LQ * DM;
  float* ssm_out = ws + o; o += (size_t)LQ * DM;
  float* qb      = ws + o; o += (size_t)LQ * DM;
  float* kbuf    = ws + o; o += (size_t)LQ * NKVH * HDIM;
  float* vbuf    = ws + o; o += (size_t)LQ * NKVH * HDIM;
  float* aob     = ws + o; o += (size_t)LQ * DM;
  float* attn_o  = ws + o; o += (size_t)LQ * DM;
  float* convb   = ws + o; o += (size_t)LQ * DM;
  float* wmb     = ws + o; o += (size_t)LQ * DM;
  float* hidden  = ws + o; o += (size_t)SEP * DHALF;
  float* dec     = ws + o; o += (size_t)SEP * DM;
  float* es      = ws + o; o += (size_t)LQ * SEP;
  float* eo      = ws + o; o += (size_t)LQ * DM;

  // 1. layernorm
  ln_kernel<<<LQ, 256, 0, stream>>>(x, ln_g, ln_b, xn);
  // 2. x-projection  xp = xn @ w_xproj
  GEMM(0, false, xn, w_xproj, nullptr, xp, LQ, XPW, DM, DM, XPW, XPW, 1.f);
  // 3. delta = softplus(dt_in @ w_dt + b_dt)   (dt_in = xp[:, :64], lda=192)
  GEMM(2, false, xp, w_dt, b_dt, delta, LQ, DM, DTRANK, XPW, DM, DM, 1.f);
  // 4. selective scan
  ssm_scan<<<DM / 256, 256, 0, stream>>>(delta, xp, xn, A_log, ys);
  // 5. ys += x * D_ssm
  addd_kernel<<<(LQ * DM) / 256, 256, 0, stream>>>(ys, x, D_ssm);
  // 6. ssm_out = ys @ w_ssm_out
  GEMM(0, false, ys, w_ssmout, nullptr, ssm_out, LQ, DM, DM, DM, DM, DM, 1.f);
  // 7. q/k/v projections
  GEMM(0, false, x, wq, nullptr, qb, LQ, DM, DM, DM, DM, DM, 1.f);
  GEMM(0, false, x, wk, nullptr, kbuf, LQ, NKVH * HDIM, DM, DM, NKVH * HDIM, NKVH * HDIM, 1.f);
  GEMM(0, false, x, wv, nullptr, vbuf, LQ, NKVH * HDIM, DM, DM, NKVH * HDIM, NKVH * HDIM, 1.f);
  // 8. sliding-window GQA attention
  attn_kernel<<<dim3(LQ / 16, NH), 128, 0, stream>>>(qb, kbuf, vbuf, rel_bias, aob);
  // 9. output projection
  GEMM(0, false, aob, wo, nullptr, attn_o, LQ, DM, DM, DM, DM, DM, 1.f);
  // 10. depthwise conv + silu
  conv_kernel<<<(LQ * DM) / 256, 256, 0, stream>>>(x, w_conv, b_conv, convb);
  // 11. wm = x @ w_wm + b_wm
  GEMM(1, false, x, w_wm, b_wm, wmb, LQ, DM, DM, DM, DM, DM, 1.f);
  // 12. hidden = gelu(episodic @ w_dc1 + b_dc1)
  GEMM(3, false, episodic, w_dc1, b_dc1, hidden, SEP, DHALF, CDIM, CDIM, DHALF, DHALF, 1.f);
  // 13. dec = hidden @ w_dc2 + b_dc2
  GEMM(1, false, hidden, w_dc2, b_dc2, dec, SEP, DM, DHALF, DHALF, DM, DM, 1.f);
  // 14. es = (x @ dec^T) / 32
  GEMM(4, true, x, dec, nullptr, es, LQ, SEP, DM, DM, DM, SEP, 0.03125f);
  // 15. row softmax
  softmax_rows<<<LQ, 256, 0, stream>>>(es);
  // 16. eo = P @ dec
  GEMM(0, false, es, dec, nullptr, eo, LQ, DM, SEP, SEP, DM, DM, 1.f);
  // 17. final combine
  combine_kernel<<<(LQ * DM) / 256, 256, 0, stream>>>(x, attn_o, ssm_out, convb, wmb, eo, out);
}
